// Head_63848983823177
// MI455X (gfx1250) — matrix operand order repaired and checked by reading, not compiled, not run on hardware
//
#include <hip/hip_runtime.h>
#include <hip/hip_bf16.h>

// ---------------- types ----------------
typedef __bf16 bf16;
typedef __attribute__((ext_vector_type(16))) __bf16 v16bf;
typedef __attribute__((ext_vector_type(8)))  __bf16 v8bf;
typedef __attribute__((ext_vector_type(4)))  __bf16 v4bf;
typedef __attribute__((ext_vector_type(8)))  float  v8f;
typedef __attribute__((ext_vector_type(4)))  int    v4i;
typedef __attribute__((address_space(1))) v4i gv4i;   // global int4
typedef __attribute__((address_space(3))) v4i lv4i;   // LDS int4

// Problem constants
#define BB   8
#define TT   2048
#define CC   512
#define HH   64
#define BT   (BB*TT)              // 16384
#define SCALE 0.044194173824159216f   // 512^-0.5

// Attention tiling
#define TILE_S 64                 // keys per iteration
#define NITER  (TT / TILE_S)      // 32
#define KSTR   72                 // padded LDS row stride (bf16 elems) for K/V
#define PSTR   72                 // padded LDS row stride for P

// Workspace layout (bytes)
#define XB_OFF  ((size_t)0)                         // x bf16 [BT, C]   16 MiB
#define WT_OFF  ((size_t)16777216)                  // 3x Wt bf16 [H,C] 3*64 KiB
#define Q_OFF   (WT_OFF + (size_t)3*65536)          // q bf16 [BT, H]   2 MiB
#define K_OFF   (Q_OFF + (size_t)2097152)           // k bf16 [BT, H]   2 MiB
#define V_OFF   (K_OFF + (size_t)2097152)           // v bf16 [B, H, T] 2 MiB

// ---------------- helpers ----------------
__device__ __forceinline__ v8f wmma_bf16(v16bf a, v16bf b, v8f c) {
    return __builtin_amdgcn_wmma_f32_16x16x32_bf16(false, a, false, b, (short)0, c,
                                                   false, false);
}

__device__ __forceinline__ v16bf combine16(v8bf lo, v8bf hi) {
    v16bf r;
#pragma unroll
    for (int e = 0; e < 8; ++e) { r[e] = lo[e]; r[e + 8] = hi[e]; }
    return r;
}

// A-operand load: 16 contiguous elements p[0..15] (two 16B chunks)
__device__ __forceinline__ v16bf load_a(const bf16* p) {
    return combine16(*(const v8bf*)p, *(const v8bf*)(p + 8));
}
// B-operand load: 16 contiguous elements as two 16B chunks (only needs 16B align)
__device__ __forceinline__ v16bf load_b(const bf16* p) {
    return combine16(*(const v8bf*)p, *(const v8bf*)(p + 8));
}

__device__ __forceinline__ float hred_max(float v) {
    v = fmaxf(v, __shfl_xor(v, 1, 32));
    v = fmaxf(v, __shfl_xor(v, 2, 32));
    v = fmaxf(v, __shfl_xor(v, 4, 32));
    v = fmaxf(v, __shfl_xor(v, 8, 32));
    return v;   // reduced within each 16-lane half
}
__device__ __forceinline__ float hred_sum(float v) {
    v += __shfl_xor(v, 1, 32);
    v += __shfl_xor(v, 2, 32);
    v += __shfl_xor(v, 4, 32);
    v += __shfl_xor(v, 8, 32);
    return v;
}

// 16-byte global -> LDS copy, async if the gfx1250 builtin is available.
__device__ __forceinline__ void copy16_g2l(bf16* lds, const bf16* g) {
#if __has_builtin(__builtin_amdgcn_global_load_async_to_lds_b128)
    __builtin_amdgcn_global_load_async_to_lds_b128(
        (gv4i*)(unsigned long long)g,
        (lv4i*)(unsigned int)(unsigned long long)lds, 0, 0);
#else
    *(v8bf*)lds = *(const v8bf*)g;
#endif
}

__device__ __forceinline__ void wait_async_lds() {
    asm volatile("s_wait_asynccnt 0x0" ::: "memory");
}

// ---------------- kernel 1: fp32 -> bf16 convert of x ----------------
__global__ __launch_bounds__(256) void cvt_x_kernel(const float* __restrict__ x,
                                                    bf16* __restrict__ xb, int n4) {
    int i = blockIdx.x * blockDim.x + threadIdx.x;
    if (i < n4) {
        float4 v = ((const float4*)x)[i];
        v4bf o;
        o[0] = (__bf16)v.x; o[1] = (__bf16)v.y; o[2] = (__bf16)v.z; o[3] = (__bf16)v.w;
        ((v4bf*)xb)[i] = o;
    }
}

// ---------------- kernel 2: transpose W [C,H] -> bf16 [H,C] ----------------
__global__ __launch_bounds__(256) void prep_w_kernel(const float* __restrict__ Wq,
                                                     const float* __restrict__ Wk,
                                                     const float* __restrict__ Wv,
                                                     bf16* __restrict__ wtq,
                                                     bf16* __restrict__ wtk,
                                                     bf16* __restrict__ wtv) {
    int i = blockIdx.x * blockDim.x + threadIdx.x;   // 0 .. 3*32768-1
    int which = i >> 15;
    int r = i & 32767;                // r = c*H + h
    int c = r >> 6;
    int h = r & 63;
    const float* W = (which == 0) ? Wq : (which == 1) ? Wk : Wv;
    bf16* wt = (which == 0) ? wtq : (which == 1) ? wtk : wtv;
    wt[(size_t)h * CC + c] = (__bf16)W[r];
}

// ---------------- kernel 3: projections q,k,v via WMMA ----------------
__global__ __launch_bounds__(128) void proj_kernel(const bf16* __restrict__ xb,
                                                   const bf16* __restrict__ wtq,
                                                   const bf16* __restrict__ wtk,
                                                   const bf16* __restrict__ wtv,
                                                   bf16* __restrict__ q,
                                                   bf16* __restrict__ k,
                                                   bf16* __restrict__ vt) {
    const int lane = threadIdx.x & 31;
    const int wave = threadIdx.x >> 5;
    const int n    = lane & 15;
    const int half = lane >> 4;
    const int t0   = blockIdx.x * 16;
    const int h0   = wave * 16;
    const int sel  = blockIdx.y;

    const bf16* wt = (sel == 0) ? wtq : (sel == 1) ? wtk : wtv;

    v8f acc = {};
    const int abase = half * 16;
    const bf16* arow = xb + (size_t)(t0 + n) * CC + abase;     // A: row = lane&15
    const bf16* brow = wt + (size_t)(h0 + n) * CC + half * 16; // B: col = lane&15

#pragma unroll
    for (int c0 = 0; c0 < CC; c0 += 32) {
        v16bf a = load_a(arow + c0);
        v16bf b = load_b(brow + c0);
        acc = wmma_bf16(a, b, acc);
    }

    if (sel == 2) {
        // v transposed: vt[b][h][t]
#pragma unroll
        for (int r = 0; r < 8; ++r) {
            int gt = t0 + r + 8 * half;
            int bb = gt >> 11;
            int tl = gt & (TT - 1);
            vt[((size_t)bb * HH + (h0 + n)) * TT + tl] = (__bf16)acc[r];
        }
    } else {
        bf16* o = (sel == 0) ? q : k;
#pragma unroll
        for (int r = 0; r < 8; ++r) {
            int gt = t0 + r + 8 * half;
            o[(size_t)gt * HH + h0 + n] = (__bf16)acc[r];
        }
    }
}

// ---------------- kernel 4: flash attention ----------------
// Block = 4 waves = 4 query tiles of 16 (same batch). 64 keys / iteration,
// K and V tiles staged into LDS with async copies, double-buffered.
__global__ __launch_bounds__(128) void attn_kernel(const bf16* __restrict__ q,
                                                   const bf16* __restrict__ kmat,
                                                   const bf16* __restrict__ vt,
                                                   float* __restrict__ out) {
    __shared__ __align__(16) bf16 kbuf[2][TILE_S * KSTR];   // [s][h], padded
    __shared__ __align__(16) bf16 vbuf[2][TILE_S * KSTR];   // [h][s], padded
    __shared__ __align__(16) bf16 pbuf[4][16 * PSTR];       // per-wave P bounce

    const int tid  = threadIdx.x;
    const int lane = tid & 31;
    const int wave = tid >> 5;
    const int n    = lane & 15;
    const int half = lane >> 4;
    const int qt   = blockIdx.x * 4 + wave;   // 0..1023
    const int b    = qt >> 7;                 // 128 query tiles per batch
    const size_t brow = (size_t)b * TT;
    const int abase = half * 16;

    // cooperative stage coordinates: each thread copies one 16B chunk per row-group
    const int srow   = tid >> 3;        // 0..15
    const int schunk = (tid & 7) * 8;   // element offset within row

    const bf16* kbatch = kmat + brow * HH;                 // [s][h]
    const bf16* vbatch = vt + (size_t)b * HH * TT;         // [h][s]

    // Q as A-operand (persistent)
    const bf16* qp = q + (size_t)(qt * 16 + n) * HH + abase;
    v16bf Q0 = load_a(qp);        // h 0..31
    v16bf Q1 = load_a(qp + 32);   // h 32..63

    v8f acc[4] = {{}, {}, {}, {}};
    float mrun[8], lrun[8];
#pragma unroll
    for (int r = 0; r < 8; ++r) { mrun[r] = -3.0e38f; lrun[r] = 0.0f; }

    bf16* myp = pbuf[wave];

    // prologue: stage tile 0 into buffer 0
#pragma unroll
    for (int g = 0; g < 4; ++g) {
        int rr = srow + 16 * g;
        copy16_g2l(&kbuf[0][rr * KSTR + schunk], kbatch + (size_t)rr * HH + schunk);
        copy16_g2l(&vbuf[0][rr * KSTR + schunk], vbatch + (size_t)rr * TT + schunk);
    }

    for (int it = 0; it < NITER; ++it) {
        const int cur = it & 1;
        wait_async_lds();
        __syncthreads();                       // current buffers ready, prev reads done

        if (it + 1 < NITER) {                  // stage next tile into other buffer
            const int s0n = (it + 1) * TILE_S;
#pragma unroll
            for (int g = 0; g < 4; ++g) {
                int rr = srow + 16 * g;
                copy16_g2l(&kbuf[cur ^ 1][rr * KSTR + schunk],
                           kbatch + (size_t)(s0n + rr) * HH + schunk);
                copy16_g2l(&vbuf[cur ^ 1][rr * KSTR + schunk],
                           vbatch + (size_t)rr * TT + s0n + schunk);
            }
        }

        // ---- scores: 4 key sub-tiles of 16, K-dim = H = 64 (2 WMMAs each) ----
        v8f S[4];
#pragma unroll
        for (int j = 0; j < 4; ++j) {
            const bf16* kb = &kbuf[cur][(16 * j + n) * KSTR + half * 16];
            v16bf K0 = load_b(kb);        // h 0..31  (lane-half selects 16)
            v16bf K1 = load_b(kb + 32);   // h 32..63
            v8f s = {};
            s = wmma_bf16(Q0, K0, s);
            s = wmma_bf16(Q1, K1, s);
            S[j] = s;
        }

        // ---- online softmax over 64 keys at once ----
#pragma unroll
        for (int r = 0; r < 8; ++r) {
            float x0 = S[0][r] * SCALE, x1 = S[1][r] * SCALE;
            float x2 = S[2][r] * SCALE, x3 = S[3][r] * SCALE;
            float rmax = hred_max(fmaxf(fmaxf(x0, x1), fmaxf(x2, x3)));
            float mnew = fmaxf(mrun[r], rmax);
            float alpha = __expf(mrun[r] - mnew);
            x0 = __expf(x0 - mnew); x1 = __expf(x1 - mnew);
            x2 = __expf(x2 - mnew); x3 = __expf(x3 - mnew);
            float rs = hred_sum(((x0 + x1) + (x2 + x3)));
            lrun[r] = lrun[r] * alpha + rs;
            mrun[r] = mnew;
            acc[0][r] *= alpha; acc[1][r] *= alpha;
            acc[2][r] *= alpha; acc[3][r] *= alpha;
            S[0][r] = x0; S[1][r] = x1; S[2][r] = x2; S[3][r] = x3;
        }

        // ---- relayout P (16x64): D-layout -> A-layout via per-wave LDS ----
#pragma unroll
        for (int r = 0; r < 8; ++r) {
            int m = r + 8 * half;
#pragma unroll
            for (int j = 0; j < 4; ++j)
                myp[m * PSTR + 16 * j + n] = (__bf16)S[j][r];
        }
        asm volatile("s_wait_dscnt 0x0" ::: "memory");   // in-wave LDS RAW fence
        v16bf P0 = load_a(myp + n * PSTR + abase);        // keys 0..31
        v16bf P1 = load_a(myp + n * PSTR + 32 + abase);   // keys 32..63

        // ---- P x V: 4 h-tiles, K-dim = 64 keys (2 WMMAs each) ----
#pragma unroll
        for (int j = 0; j < 4; ++j) {
            const bf16* vb = &vbuf[cur][(16 * j + n) * KSTR + half * 16];
            v16bf V0 = load_b(vb);        // keys 0..31
            v16bf V1 = load_b(vb + 32);   // keys 32..63
            acc[j] = wmma_bf16(P0, V0, acc[j]);
            acc[j] = wmma_bf16(P1, V1, acc[j]);
        }
    }

    // epilogue: divide by running sum, store fp32 (coalesced across lanes)
#pragma unroll
    for (int r = 0; r < 8; ++r) {
        float inv = 1.0f / lrun[r];
        size_t row = (size_t)(qt * 16 + r + 8 * half) * HH;
        out[row + n]      = acc[0][r] * inv;
        out[row + 16 + n] = acc[1][r] * inv;
        out[row + 32 + n] = acc[2][r] * inv;
        out[row + 48 + n] = acc[3][r] * inv;
    }
}

// ---------------- launcher ----------------
extern "C" void kernel_launch(void* const* d_in, const int* in_sizes, int n_in,
                              void* d_out, int out_size, void* d_ws, size_t ws_size,
                              hipStream_t stream) {
    const float* x  = (const float*)d_in[0];
    const float* Wq = (const float*)d_in[1];
    const float* Wk = (const float*)d_in[2];
    const float* Wv = (const float*)d_in[3];
    float* out = (float*)d_out;

    char* ws = (char*)d_ws;
    bf16* xb  = (bf16*)(ws + XB_OFF);
    bf16* wtq = (bf16*)(ws + WT_OFF);
    bf16* wtk = wtq + 32768;
    bf16* wtv = wtk + 32768;
    bf16* qb  = (bf16*)(ws + Q_OFF);
    bf16* kb  = (bf16*)(ws + K_OFF);
    bf16* vtb = (bf16*)(ws + V_OFF);

    cvt_x_kernel<<<8192, 256, 0, stream>>>(x, xb, BT * CC / 4);
    prep_w_kernel<<<384, 256, 0, stream>>>(Wq, Wk, Wv, wtq, wtk, wtv);
    proj_kernel<<<dim3(BT / 16, 3), 128, 0, stream>>>(xb, wtq, wtk, wtv, qb, kb, vtb);
    attn_kernel<<<BT / 16 / 4, 128, 0, stream>>>(qb, kb, vtb, out);
}
